// WaveNet_61108794688029
// MI455X (gfx1250) — matrix operand order repaired and checked by reading, not compiled, not run on hardware
//
#include <hip/hip_runtime.h>

typedef _Float16 h16;
typedef __attribute__((ext_vector_type(16))) _Float16 v16h;
typedef __attribute__((ext_vector_type(8)))  float    v8f;

#define LSTRIDE 136            // h16 elements per padded LDS row (272B = 16B aligned)
#define LSTRIDE_U32 68

__device__ __forceinline__ v8f wmma_f16(v16h a, v16h b, v8f c) {
  return __builtin_amdgcn_wmma_f32_16x16x32_f16(false, a, false, b, (short)0, c, false, false);
}

// A fragment (16x32 f16, M x K): W row-major [M][LSTRIDE], tile at (m0, k0)
__device__ __forceinline__ v16h load_a(const h16* W, int m0, int k0, int lane) {
  const int hlf = lane >> 4;
  const h16* row = W + (m0 + (lane & 15)) * LSTRIDE + k0 + 16 * hlf;
  v16h a;
#pragma unroll
  for (int j = 0; j < 4; ++j) { a[2*j] = row[2*j]; a[2*j+1] = row[2*j+1]; }
  const h16* row2 = row + 8;
#pragma unroll
  for (int j = 0; j < 4; ++j) { a[8+2*j] = row2[2*j]; a[8+2*j+1] = row2[2*j+1]; }
  return a;
}

// B fragment (32x16 f16, K x N): X stored transposed [N][LSTRIDE] (row n holds K)
__device__ __forceinline__ v16h load_b(const h16* X, int n0, int k0, int lane) {
  const int hlf = lane >> 4;
  const h16* row = X + (n0 + (lane & 15)) * LSTRIDE + k0 + 16 * hlf;
  v16h b;
#pragma unroll
  for (int e = 0; e < 16; ++e) b[e] = row[e];
  return b;
}

// ---------------- prep kernels ----------------

// src: [10][128][128][2] fp32  ->  dst: [10][2][128][128] f16
__global__ void k_tap2(const float* __restrict__ src, h16* __restrict__ dst) {
  int idx = blockIdx.x * 256 + threadIdx.x;           // total 10*2*128*128 = 327680
  int k = idx & 127, m = (idx >> 7) & 127, t = (idx >> 14) & 1, i = idx >> 15;
  dst[idx] = (h16)src[(((i * 128 + m) * 128 + k) * 2) + t];
}

__global__ void k_plain(const float* __restrict__ src, h16* __restrict__ dst, int total) {
  int idx = blockIdx.x * 256 + threadIdx.x;
  if (idx < total) dst[idx] = (h16)src[idx];
}

// x[b][r][l] = start_w[r] * in[b][0][l]
__global__ void k_start(const float* __restrict__ in, const float* __restrict__ sw,
                        float* __restrict__ x, int L) {
  int idx = blockIdx.x * 256 + threadIdx.x;           // total 4*128*L
  int l = idx % L; int r = (idx / L) & 127; int b = idx / (L * 128);
  x[idx] = sw[r] * in[b * L + l];
}

__global__ void k_zero(float* __restrict__ p, long n) {
  long i = (long)blockIdx.x * 256 + threadIdx.x;
  if (i < n) p[i] = 0.f;
}

// ---------------- dilated conv: h = Wd0 * xp_l + Wd1 * xp_{l+d} ----------------
__global__ void k_dilated(const float* __restrict__ x, const h16* __restrict__ wd,
                          h16* __restrict__ h, int Lin, int Lh, int d) {
  extern __shared__ char smem_raw[];
  h16* W0 = (h16*)smem_raw;                 // [128][LSTRIDE]
  h16* W1 = W0 + 128 * LSTRIDE;
  h16* X0 = W1 + 128 * LSTRIDE;             // [32][LSTRIDE]  tap0 slab
  h16* X1 = X0 + 32 * LSTRIDE;              // [32][LSTRIDE]  tap1 slab
  const int tid = threadIdx.x;
  const int b = blockIdx.y;
  const int q0 = blockIdx.x * 32;

  // stage weights (already f16, [2][128][128]) into padded LDS
  const uint32_t* wsrc = (const uint32_t*)wd;
  uint32_t* W0u = (uint32_t*)W0;
  uint32_t* W1u = (uint32_t*)W1;
  for (int idx = tid; idx < 128 * 64; idx += 256) {
    int m = idx >> 6, kp = idx & 63;
    W0u[m * LSTRIDE_U32 + kp] = wsrc[idx];
    W1u[m * LSTRIDE_U32 + kp] = wsrc[128 * 64 + idx];
  }
  // stage two shifted activation slabs, transposed [n][k], fp32->f16
  for (int idx = tid; idx < 32 * 128; idx += 256) {
    int n = idx & 31, k = idx >> 5;
    long base = ((long)b * 128 + k) * Lin;
    int p0 = q0 + n - 1;          // padding=1
    int p1 = p0 + d;
    float v0 = (p0 >= 0 && p0 < Lin) ? x[base + p0] : 0.f;
    float v1 = (p1 >= 0 && p1 < Lin) ? x[base + p1] : 0.f;
    X0[n * LSTRIDE + k] = (h16)v0;
    X1[n * LSTRIDE + k] = (h16)v1;
  }
  __syncthreads();

  const int wave = tid >> 5, lane = tid & 31;
  const int m0 = wave * 16;
  v8f c0 = {0,0,0,0,0,0,0,0}, c1 = {0,0,0,0,0,0,0,0};
#pragma unroll
  for (int k0 = 0; k0 < 128; k0 += 32) {
    v16h a0  = load_a(W0, m0, k0, lane);
    v16h a1  = load_a(W1, m0, k0, lane);
    v16h b00 = load_b(X0, 0,  k0, lane);
    v16h b01 = load_b(X0, 16, k0, lane);
    v16h b10 = load_b(X1, 0,  k0, lane);
    v16h b11 = load_b(X1, 16, k0, lane);
    c0 = wmma_f16(a0, b00, c0);
    c0 = wmma_f16(a1, b10, c0);
    c1 = wmma_f16(a0, b01, c1);
    c1 = wmma_f16(a1, b11, c1);
  }
  const int hlf = lane >> 4, nn = lane & 15;
#pragma unroll
  for (int r = 0; r < 8; ++r) {
    int m = m0 + r + 8 * hlf;
    long row = ((long)b * 128 + m) * Lh;
    int p = q0 + nn;
    if (p < Lh) h[row + p] = (h16)c0[r];
    p = q0 + 16 + nn;
    if (p < Lh) h[row + p] = (h16)c1[r];
  }
}

// ---- gated unit + skip + residual:
//   z = tanh(Wf0 h_l + Wf1 h_{l+1}) * sigmoid(Wg0 h_l + Wg1 h_{l+1})
//   skip += Ws z (end-aligned); x_out = z + Wr z
__global__ void k_gate(const h16* __restrict__ h, const h16* __restrict__ wf,
                       const h16* __restrict__ wg, const h16* __restrict__ wskip,
                       const h16* __restrict__ wres, float* __restrict__ xout,
                       float* __restrict__ skipacc, int Lh, int Lz, int skip_off, int Lskip) {
  extern __shared__ char smem_raw[];
  h16* WF0 = (h16*)smem_raw;                // 4 weight matrices [128][LSTRIDE]
  h16* WF1 = WF0 + 128 * LSTRIDE;
  h16* WG0 = WF1 + 128 * LSTRIDE;
  h16* WG1 = WG0 + 128 * LSTRIDE;
  h16* Ht  = WG1 + 128 * LSTRIDE;           // [34][LSTRIDE] (33 cols + pad)
  h16* Zt  = Ht + 34 * LSTRIDE;             // [32][LSTRIDE]
  const int tid = threadIdx.x;
  const int b = blockIdx.y;
  const int q0 = blockIdx.x * 32;

  const uint32_t* wfu = (const uint32_t*)wf;
  const uint32_t* wgu = (const uint32_t*)wg;
  uint32_t* WF0u = (uint32_t*)WF0; uint32_t* WF1u = (uint32_t*)WF1;
  uint32_t* WG0u = (uint32_t*)WG0; uint32_t* WG1u = (uint32_t*)WG1;
  for (int idx = tid; idx < 128 * 64; idx += 256) {
    int m = idx >> 6, kp = idx & 63;
    WF0u[m * LSTRIDE_U32 + kp] = wfu[idx];
    WF1u[m * LSTRIDE_U32 + kp] = wfu[128 * 64 + idx];
    WG0u[m * LSTRIDE_U32 + kp] = wgu[idx];
    WG1u[m * LSTRIDE_U32 + kp] = wgu[128 * 64 + idx];
  }
  // h tile, 33 columns (halo +1), transposed [n][k]
  for (int idx = tid; idx < 33 * 128; idx += 256) {
    int n = idx % 33, k = idx / 33;
    int p = q0 + n;
    h16 v = (h16)0.f;
    if (p < Lh) v = h[((long)b * 128 + k) * Lh + p];
    Ht[n * LSTRIDE + k] = v;
  }
  __syncthreads();

  const int wave = tid >> 5, lane = tid & 31;
  const int m0 = wave * 16;
  v8f cf0 = {0,0,0,0,0,0,0,0}, cf1 = {0,0,0,0,0,0,0,0};
  v8f cg0 = {0,0,0,0,0,0,0,0}, cg1 = {0,0,0,0,0,0,0,0};
#pragma unroll
  for (int k0 = 0; k0 < 128; k0 += 32) {
    v16h af0 = load_a(WF0, m0, k0, lane);
    v16h af1 = load_a(WF1, m0, k0, lane);
    v16h ag0 = load_a(WG0, m0, k0, lane);
    v16h ag1 = load_a(WG1, m0, k0, lane);
    v16h b0  = load_b(Ht, 0,  k0, lane);
    v16h b1  = load_b(Ht, 1,  k0, lane);
    v16h b16 = load_b(Ht, 16, k0, lane);
    v16h b17 = load_b(Ht, 17, k0, lane);
    cf0 = wmma_f16(af0, b0,  cf0); cf0 = wmma_f16(af1, b1,  cf0);
    cf1 = wmma_f16(af0, b16, cf1); cf1 = wmma_f16(af1, b17, cf1);
    cg0 = wmma_f16(ag0, b0,  cg0); cg0 = wmma_f16(ag1, b1,  cg0);
    cg1 = wmma_f16(ag0, b16, cg1); cg1 = wmma_f16(ag1, b17, cg1);
  }
  float z0[8], z1[8];
#pragma unroll
  for (int r = 0; r < 8; ++r) {
    z0[r] = tanhf(cf0[r]) * (1.f / (1.f + __expf(-cg0[r])));
    z1[r] = tanhf(cf1[r]) * (1.f / (1.f + __expf(-cg1[r])));
  }
  const int hlf = lane >> 4, nn = lane & 15;
#pragma unroll
  for (int r = 0; r < 8; ++r) {
    int m = m0 + r + 8 * hlf;
    Zt[nn * LSTRIDE + m] = (h16)z0[r];
    Zt[(16 + nn) * LSTRIDE + m] = (h16)z1[r];
  }
  __syncthreads();
  // reuse WF0/WF1 space for skip & residual weights
  const uint32_t* wsu = (const uint32_t*)wskip;
  const uint32_t* wru = (const uint32_t*)wres;
  for (int idx = tid; idx < 128 * 64; idx += 256) {
    int m = idx >> 6, kp = idx & 63;
    WF0u[m * LSTRIDE_U32 + kp] = wsu[idx];
    WF1u[m * LSTRIDE_U32 + kp] = wru[idx];
  }
  __syncthreads();

  v8f cs0 = {0,0,0,0,0,0,0,0}, cs1 = {0,0,0,0,0,0,0,0};
  v8f cr0 = {0,0,0,0,0,0,0,0}, cr1 = {0,0,0,0,0,0,0,0};
#pragma unroll
  for (int k0 = 0; k0 < 128; k0 += 32) {
    v16h as  = load_a(WF0, m0, k0, lane);
    v16h ar  = load_a(WF1, m0, k0, lane);
    v16h bz0 = load_b(Zt, 0,  k0, lane);
    v16h bz1 = load_b(Zt, 16, k0, lane);
    cs0 = wmma_f16(as, bz0, cs0); cs1 = wmma_f16(as, bz1, cs1);
    cr0 = wmma_f16(ar, bz0, cr0); cr1 = wmma_f16(ar, bz1, cr1);
  }
#pragma unroll
  for (int r = 0; r < 8; ++r) {
    int m = m0 + r + 8 * hlf;
    long xrow = ((long)b * 128 + m) * Lz;
    long srow = ((long)b * 128 + m) * Lskip;
    int p = q0 + nn;
    if (p < Lz) {
      xout[xrow + p] = z0[r] + cr0[r];
      int ps = p - skip_off;
      if (ps >= 0) skipacc[srow + ps] += cs0[r];
    }
    p = q0 + 16 + nn;
    if (p < Lz) {
      xout[xrow + p] = z1[r] + cr1[r];
      int ps = p - skip_off;
      if (ps >= 0) skipacc[srow + ps] += cs1[r];
    }
  }
}

// ---- end: out = E2 * relu(E1 * relu(skip) + b1) + b2 ----
__global__ void k_end(const float* __restrict__ skipacc, const h16* __restrict__ e1,
                      const h16* __restrict__ e2, const float* __restrict__ bias1,
                      const float* __restrict__ bias2, float* __restrict__ out, int Lk) {
  extern __shared__ char smem_raw[];
  h16* E1 = (h16*)smem_raw;                 // [128][LSTRIDE]
  h16* E2 = E1 + 128 * LSTRIDE;             // [256][LSTRIDE]
  h16* St = E2 + 256 * LSTRIDE;             // [32][LSTRIDE]
  h16* Yt = St + 32 * LSTRIDE;              // [32][LSTRIDE]
  const int tid = threadIdx.x;
  const int b = blockIdx.y;
  const int q0 = blockIdx.x * 32;

  const uint32_t* e1u = (const uint32_t*)e1;
  const uint32_t* e2u = (const uint32_t*)e2;
  uint32_t* E1u = (uint32_t*)E1; uint32_t* E2u = (uint32_t*)E2;
  for (int idx = tid; idx < 128 * 64; idx += 256) {
    int m = idx >> 6, kp = idx & 63;
    E1u[m * LSTRIDE_U32 + kp] = e1u[idx];
  }
  for (int idx = tid; idx < 256 * 64; idx += 256) {
    int m = idx >> 6, kp = idx & 63;
    E2u[m * LSTRIDE_U32 + kp] = e2u[idx];
  }
  for (int idx = tid; idx < 32 * 128; idx += 256) {
    int n = idx & 31, k = idx >> 5;
    int p = q0 + n;
    float v = 0.f;
    if (p < Lk) { v = skipacc[((long)b * 128 + k) * Lk + p]; v = v > 0.f ? v : 0.f; }
    St[n * LSTRIDE + k] = (h16)v;
  }
  __syncthreads();

  const int wave = tid >> 5, lane = tid & 31;
  const int hlf = lane >> 4, nn = lane & 15;
  // stage 1: M=128
  {
    const int m0 = wave * 16;
    v8f c0 = {0,0,0,0,0,0,0,0}, c1 = {0,0,0,0,0,0,0,0};
#pragma unroll
    for (int k0 = 0; k0 < 128; k0 += 32) {
      v16h a  = load_a(E1, m0, k0, lane);
      v16h b0 = load_b(St, 0,  k0, lane);
      v16h b1 = load_b(St, 16, k0, lane);
      c0 = wmma_f16(a, b0, c0);
      c1 = wmma_f16(a, b1, c1);
    }
#pragma unroll
    for (int r = 0; r < 8; ++r) {
      int m = m0 + r + 8 * hlf;
      float bb = bias1[m];
      float y0 = c0[r] + bb; y0 = y0 > 0.f ? y0 : 0.f;
      float y1 = c1[r] + bb; y1 = y1 > 0.f ? y1 : 0.f;
      Yt[nn * LSTRIDE + m] = (h16)y0;
      Yt[(16 + nn) * LSTRIDE + m] = (h16)y1;
    }
  }
  __syncthreads();
  // stage 2: M=256, each wave owns rows [wave*32, wave*32+32)
  {
    const int m0 = wave * 32;
    v8f c00 = {0,0,0,0,0,0,0,0}, c01 = {0,0,0,0,0,0,0,0};
    v8f c10 = {0,0,0,0,0,0,0,0}, c11 = {0,0,0,0,0,0,0,0};
#pragma unroll
    for (int k0 = 0; k0 < 128; k0 += 32) {
      v16h a0  = load_a(E2, m0,      k0, lane);
      v16h a1  = load_a(E2, m0 + 16, k0, lane);
      v16h bz0 = load_b(Yt, 0,  k0, lane);
      v16h bz1 = load_b(Yt, 16, k0, lane);
      c00 = wmma_f16(a0, bz0, c00); c01 = wmma_f16(a0, bz1, c01);
      c10 = wmma_f16(a1, bz0, c10); c11 = wmma_f16(a1, bz1, c11);
    }
#pragma unroll
    for (int r = 0; r < 8; ++r) {
      int mA = m0 + r + 8 * hlf;
      int mB = mA + 16;
      long rowA = ((long)b * 256 + mA) * Lk;
      long rowB = ((long)b * 256 + mB) * Lk;
      int p = q0 + nn;
      if (p < Lk) { out[rowA + p] = c00[r] + bias2[mA]; out[rowB + p] = c10[r] + bias2[mB]; }
      p = q0 + 16 + nn;
      if (p < Lk) { out[rowA + p] = c01[r] + bias2[mA]; out[rowB + p] = c11[r] + bias2[mB]; }
    }
  }
}

// ---------------- host ----------------
extern "C" void kernel_launch(void* const* d_in, const int* in_sizes, int n_in,
                              void* d_out, int out_size, void* d_ws, size_t ws_size,
                              hipStream_t stream) {
  const float* in_x = (const float*)d_in[0];
  const float* sw   = (const float*)d_in[1];
  const float* dw   = (const float*)d_in[2];
  const float* fw   = (const float*)d_in[3];
  const float* gw   = (const float*)d_in[4];
  const float* rw   = (const float*)d_in[5];
  const float* skw  = (const float*)d_in[6];
  const float* e1w  = (const float*)d_in[7];
  const float* e1b  = (const float*)d_in[8];
  const float* e2w  = (const float*)d_in[9];
  const float* e2b  = (const float*)d_in[10];
  float* out = (float*)d_out;

  char* base = (char*)d_ws;
  size_t off = 0;
  auto carve = [&](size_t bytes) -> void* {
    void* p = base + off;
    off += (bytes + 255) & ~(size_t)255;
    return p;
  };
  const int Lskip = 15371;
  float* xA  = (float*)carve((size_t)4 * 128 * 16384 * 4);
  float* xB  = (float*)carve((size_t)4 * 128 * 16384 * 4);
  h16*  hbuf = (h16*) carve((size_t)4 * 128 * 16385 * 2);
  float* skp = (float*)carve((size_t)4 * 128 * Lskip * 4);
  h16*  wdh  = (h16*) carve((size_t)10 * 2 * 128 * 128 * 2);
  h16*  wfh  = (h16*) carve((size_t)10 * 2 * 128 * 128 * 2);
  h16*  wgh  = (h16*) carve((size_t)10 * 2 * 128 * 128 * 2);
  h16*  wrh  = (h16*) carve((size_t)10 * 128 * 128 * 2);
  h16*  wsh  = (h16*) carve((size_t)10 * 128 * 128 * 2);
  h16*  e1h  = (h16*) carve((size_t)128 * 128 * 2);
  h16*  e2h  = (h16*) carve((size_t)256 * 128 * 2);

  const size_t SH_H = (size_t)(2 * 128 + 2 * 32) * LSTRIDE * 2;           // 87,040 B
  const size_t SH_Z = (size_t)(4 * 128 + 34 + 32) * LSTRIDE * 2;          // 157,216 B
  const size_t SH_E = (size_t)(128 + 256 + 32 + 32) * LSTRIDE * 2;        // 121,856 B
  hipFuncSetAttribute((const void*)k_dilated, hipFuncAttributeMaxDynamicSharedMemorySize, (int)SH_H);
  hipFuncSetAttribute((const void*)k_gate,    hipFuncAttributeMaxDynamicSharedMemorySize, (int)SH_Z);
  hipFuncSetAttribute((const void*)k_end,     hipFuncAttributeMaxDynamicSharedMemorySize, (int)SH_E);

  // weight prep (fp32 -> f16, tap-major for k=2 convs)
  k_tap2<<<1280, 256, 0, stream>>>(dw, wdh);
  k_tap2<<<1280, 256, 0, stream>>>(fw, wfh);
  k_tap2<<<1280, 256, 0, stream>>>(gw, wgh);
  k_plain<<<640, 256, 0, stream>>>(rw,  wrh, 10 * 128 * 128);
  k_plain<<<640, 256, 0, stream>>>(skw, wsh, 10 * 128 * 128);
  k_plain<<<64,  256, 0, stream>>>(e1w, e1h, 128 * 128);
  k_plain<<<128, 256, 0, stream>>>(e2w, e2h, 256 * 128);

  k_start<<<(4 * 128 * 16384) / 256, 256, 0, stream>>>(in_x, sw, xA, 16384);
  {
    long n = (long)4 * 128 * Lskip;
    k_zero<<<(unsigned)((n + 255) / 256), 256, 0, stream>>>(skp, n);
  }

  int Lcur = 16384;
  float* xin = xA;
  float* xot = xB;
  for (int i = 0; i < 10; ++i) {
    int d  = 1 << i;
    int Lh = Lcur + 2 - d;
    int Lz = Lh - 1;
    dim3 gh((unsigned)((Lh + 31) / 32), 4);
    k_dilated<<<gh, 256, SH_H, stream>>>(xin, wdh + (size_t)i * 2 * 16384, hbuf, Lcur, Lh, d);
    dim3 gz((unsigned)((Lz + 31) / 32), 4);
    k_gate<<<gz, 256, SH_Z, stream>>>(hbuf, wfh + (size_t)i * 2 * 16384, wgh + (size_t)i * 2 * 16384,
                                      wsh + (size_t)i * 16384, wrh + (size_t)i * 16384,
                                      xot, skp, Lh, Lz, Lz - Lskip, Lskip);
    float* t = xin; xin = xot; xot = t;
    Lcur = Lz;
  }
  dim3 ge((unsigned)((Lskip + 31) / 32), 4);
  k_end<<<ge, 256, SH_E, stream>>>(skp, e1h, e2h, e1b, e2b, out, Lskip);
}
